// TransformersEncoder_62388694942467
// MI455X (gfx1250) — matrix operand order repaired and checked by reading, not compiled, not run on hardware
//
#include <hip/hip_runtime.h>
#include <hip/hip_bf16.h>
#include <math.h>

// ---------------------------------------------------------------------------
// DeBERTa-v2 encoder on MI455X (gfx1250, wave32, WMMA).
// All GEMMs run on v_wmma_f32_16x16x32_f16 (f16 in, f32 accumulate).
// Compute-bound workload (~72 GFLOP vs ~30MB traffic) -> WMMA everywhere.
// v3: 4x M register blocking per wave (B fragment reused by 4 WMMAs) with
//     all fragments preloaded per K-step so loads clause together and waits
//     stage (0x6..0x0) instead of a full drain before each WMMA.
// ---------------------------------------------------------------------------

typedef _Float16 f16;
typedef _Float16 h8_t  __attribute__((ext_vector_type(8)));
typedef _Float16 h16_t __attribute__((ext_vector_type(16)));
typedef float    f8_t  __attribute__((ext_vector_type(8)));

#define KB     8
#define KS     512
#define KHID   512
#define KHEADS 8
#define KDH    64
#define KFF    768
#define KNL    3
#define KSPAN  256
#define KMAXP  512
#define KVOCAB 100

// ---------------------------------------------------------------------------
// f32 -> f16 conversion (weights)
// ---------------------------------------------------------------------------
__global__ void cvt_f16_kernel(const float* __restrict__ x, f16* __restrict__ y,
                               long long n) {
  long long i = (long long)blockIdx.x * blockDim.x + threadIdx.x;
  if (i < n) y[i] = (f16)x[i];
}

// ---------------------------------------------------------------------------
// DeBERTa-v2 log-bucket relative position -> gather index tables.
//   rel[i,j] = i - j ; bucketed ; c2p_idx = clip(bucket+SPAN), p2c_idx = clip(-bucket+SPAN)
// ---------------------------------------------------------------------------
__global__ void relidx_kernel(int* __restrict__ c2pI, int* __restrict__ p2cI) {
  int idx = blockIdx.x * blockDim.x + threadIdx.x;
  if (idx >= KS * KS) return;
  int i = idx / KS, j = idx % KS;
  int rel = i - j;
  const int mid = KSPAN / 2;  // 128
  int arel = rel < 0 ? -rel : rel;
  double absp = (rel < mid && rel > -mid) ? (double)(mid - 1) : (double)arel;
  double logp =
      ceil(log(absp / (double)mid) / log((double)(KMAXP - 1) / (double)mid) *
           (double)(mid - 1)) + (double)mid;
  double sgn = rel > 0 ? 1.0 : (rel < 0 ? -1.0 : 0.0);
  double bucket = (absp <= (double)mid) ? (double)rel : logp * sgn;
  int bp = (int)bucket;
  int c = bp + KSPAN;  c = c < 0 ? 0 : (c > 2 * KSPAN - 1 ? 2 * KSPAN - 1 : c);
  int p = -bp + KSPAN; p = p < 0 ? 0 : (p > 2 * KSPAN - 1 ? 2 * KSPAN - 1 : p);
  c2pI[idx] = c;
  p2cI[idx] = p;
}

// ---------------------------------------------------------------------------
// Row LayerNorm, optional residual add, optional per-row mask multiply.
// Writes f32 and/or f16. One block per row (cols <= 512 here).
// ---------------------------------------------------------------------------
__global__ __launch_bounds__(256)
void ln_kernel(const float* __restrict__ X, const float* __restrict__ R,
               const float* __restrict__ g, const float* __restrict__ bta,
               const float* __restrict__ rowmask, int cols,
               float* __restrict__ Yf, f16* __restrict__ Yh) {
  int row = blockIdx.x;
  const float* x = X + (long long)row * cols;
  const float* r = R ? R + (long long)row * cols : nullptr;
  __shared__ float s1[256], s2[256];
  float lsum = 0.f, lsq = 0.f;
  for (int c = threadIdx.x; c < cols; c += 256) {
    float v = x[c] + (r ? r[c] : 0.f);
    lsum += v; lsq += v * v;
  }
  s1[threadIdx.x] = lsum; s2[threadIdx.x] = lsq;
  __syncthreads();
  for (int st = 128; st > 0; st >>= 1) {
    if (threadIdx.x < st) {
      s1[threadIdx.x] += s1[threadIdx.x + st];
      s2[threadIdx.x] += s2[threadIdx.x + st];
    }
    __syncthreads();
  }
  float mu  = s1[0] / cols;
  float var = s2[0] / cols - mu * mu;
  float inv = rsqrtf(var + 1e-7f);
  float mk  = rowmask ? rowmask[row] : 1.f;
  for (int c = threadIdx.x; c < cols; c += 256) {
    float v = x[c] + (r ? r[c] : 0.f);
    float y = ((v - mu) * inv * g[c] + bta[c]) * mk;
    if (Yf) Yf[(long long)row * cols + c] = y;
    if (Yh) Yh[(long long)row * cols + c] = (f16)y;
  }
}

// ---------------------------------------------------------------------------
// Generic batched WMMA GEMM:  C = alpha * (A @ B(^T)) + bias, optional GELU.
// A: f16 row-major [M,K]   (batch offset zo*sAo + zi*sAi, zo=z/inner, zi=z%inner)
// B: f16 row-major [K,N]   (or [N,K] if TRANSB)
// C: f32 and/or f16 outputs, guarded at M/N edges (N=100 classifier).
//
// Tiling: block = 128 threads = 4 waves -> 64(M) x 64(N) C tile.
//   wave w owns columns n0 = blockIdx.y*64 + w*16 and ALL 64 rows:
//   4 stacked 16x16 accumulators; per 32-wide K step the B fragment is
//   loaded once and reused by 4 WMMAs (4x matrix-op density per byte).
//   All fragments are preloaded before the WMMA block so the loads form one
//   s_clause and the waits stage down instead of draining per WMMA.
//
// Fragment layouts used here (wave32; A/B k order is contiguous, not the ISA table order):
//   A elem e, lane L: m=L%16, k = e + (L>=16?16:0)
//   B elem e, lane L: n=L%16, k = e + (L>=16?16:0)
//   C elem r, lane L: n=L%16, m = r + (L>=16?8:0)
// ---------------------------------------------------------------------------
template <bool TRANSB>
__global__ __launch_bounds__(128)
void gemm_wmma(const f16* __restrict__ A, const f16* __restrict__ Bm,
               const float* __restrict__ bias,
               float* __restrict__ Cf, f16* __restrict__ Ch,
               int M, int N, int K, int lda, int ldb, int ldc,
               long long sAo, long long sAi, long long sBo, long long sBi,
               long long sCo, long long sCi, int innerCnt,
               float alpha, int act) {
  int z  = blockIdx.z;
  int zo = z / innerCnt, zi = z % innerCnt;
  const f16* Ab = A + (long long)zo * sAo + (long long)zi * sAi;
  const f16* Bb = Bm + (long long)zo * sBo + (long long)zi * sBi;
  long long coff = (long long)zo * sCo + (long long)zi * sCi;

  int wave = threadIdx.x >> 5;
  int lane = threadIdx.x & 31;
  int m0 = blockIdx.x * 64;
  int n0 = blockIdx.y * 64 + wave * 16;

  int hi = lane >> 4;            // 0 / 1 half-wave
  int mr = lane & 15;
  int nc = n0 + mr;
  bool nok = nc < N;

  // Row pointers for the 4 M-subtiles (clamped: M is a multiple of 64 for all
  // call sites in this network; clamp keeps stray rows in-bounds regardless).
  const f16* Arow[4];
#pragma unroll
  for (int t = 0; t < 4; ++t) {
    int mm = m0 + t * 16 + mr;
    if (mm >= M) mm = M - 1;
    Arow[t] = Ab + (long long)mm * lda;
  }

  f8_t acc[4] = {};
  for (int kb = 0; kb < K; kb += 32) {
    // ---- preload B fragment (shared by all 4 WMMAs this K-step)
    h16_t bfr = {};
    if (TRANSB) {
      if (nok) {
        const f16* Brow = Bb + (long long)nc * ldb + kb + hi * 16;
        h8_t blo = *(const h8_t*)(Brow);
        h8_t bhi = *(const h8_t*)(Brow + 8);
        if (kb + 32 < K) __builtin_prefetch(Brow + 32, 0, 1);  // global_prefetch_b8
#pragma unroll
        for (int e = 0; e < 8; ++e) { bfr[e] = blo[e]; bfr[e + 8] = bhi[e]; }
      }
    } else {
      if (nok) {
#pragma unroll
        for (int e = 0; e < 16; ++e)
          bfr[e] = Bb[(long long)(kb + hi * 16 + e) * ldb + nc];
      }
    }

    // ---- preload ALL 4 A fragments (one clause of b128 loads)
    h16_t af[4];
#pragma unroll
    for (int t = 0; t < 4; ++t) {
      h8_t alo = *(const h8_t*)(Arow[t] + kb + hi * 16);
      h8_t ahi = *(const h8_t*)(Arow[t] + kb + hi * 16 + 8);
#pragma unroll
      for (int e = 0; e < 8; ++e) { af[t][e] = alo[e]; af[t][e + 8] = ahi[e]; }
    }
    if (kb + 32 < K) __builtin_prefetch(Arow[0] + kb + 32, 0, 1);

    // ---- 4 independent WMMA issues (hazard window covered by independence)
#pragma unroll
    for (int t = 0; t < 4; ++t) {
      acc[t] = __builtin_amdgcn_wmma_f32_16x16x32_f16(
          /*neg_a=*/false, af[t], /*neg_b=*/false, bfr,
          /*c_mod=*/(short)0, acc[t], /*reuse_a=*/false, /*reuse_b=*/false);
    }
  }

  // ---- epilogue: alpha, bias, optional exact-erf GELU, guarded stores
#pragma unroll
  for (int t = 0; t < 4; ++t) {
#pragma unroll
    for (int r = 0; r < 8; ++r) {
      int mm = m0 + t * 16 + hi * 8 + r;
      if (mm < M && nok) {
        float v = alpha * acc[t][r];
        if (bias) v += bias[nc];
        if (act == 1) v = 0.5f * v * (1.0f + erff(v * 0.70710678118654752f));
        long long o = coff + (long long)mm * ldc + nc;
        if (Cf) Cf[o] = v;
        if (Ch) Ch[o] = (f16)v;
      }
    }
  }
}

// ---------------------------------------------------------------------------
// Gather c2p/p2c terms into scores, masked softmax (XSoftmax), write f16 probs.
// One block per (b,h,q) row; 256 threads * 2 keys (S=512).
// ---------------------------------------------------------------------------
__global__ __launch_bounds__(256)
void softmax_gather_kernel(const float* __restrict__ scores,
                           const float* __restrict__ c2p,
                           const float* __restrict__ p2c,
                           const int* __restrict__ c2pI,
                           const int* __restrict__ p2cI,
                           const float* __restrict__ amask,
                           f16* __restrict__ probs) {
  int q  = blockIdx.x % KS;
  int bh = blockIdx.x / KS;
  int b  = bh / KHEADS;
  long long rowo = ((long long)bh * KS + q) * KS;
  const float* sc = scores + rowo;
  const float* cr = c2p + rowo;
  const float* pb = p2c + (long long)bh * KS * KS;
  bool mq = amask[b * KS + q] > 0.f;

  __shared__ float red[256];
  float vals[2]; bool ok[2];
  float lmax = -3.402823466e38f;
#pragma unroll
  for (int t = 0; t < 2; ++t) {
    int k = threadIdx.x + t * 256;
    float s = sc[k] + cr[c2pI[q * KS + k]] +
              pb[(long long)k * KS + p2cI[k * KS + q]];
    ok[t] = mq && (amask[b * KS + k] > 0.f);
    vals[t] = ok[t] ? s : -3.402823466e38f;
    lmax = fmaxf(lmax, vals[t]);
  }
  red[threadIdx.x] = lmax; __syncthreads();
  for (int st = 128; st > 0; st >>= 1) {
    if (threadIdx.x < st)
      red[threadIdx.x] = fmaxf(red[threadIdx.x], red[threadIdx.x + st]);
    __syncthreads();
  }
  float mx = red[0]; __syncthreads();
  float e0 = __expf(vals[0] - mx), e1 = __expf(vals[1] - mx);
  red[threadIdx.x] = e0 + e1; __syncthreads();
  for (int st = 128; st > 0; st >>= 1) {
    if (threadIdx.x < st) red[threadIdx.x] += red[threadIdx.x + st];
    __syncthreads();
  }
  float inv = 1.f / red[0];
  f16* pr = probs + rowo;
  pr[threadIdx.x]       = (f16)(ok[0] ? e0 * inv : 0.f);
  pr[threadIdx.x + 256] = (f16)(ok[1] ? e1 * inv : 0.f);
}

// ---------------------------------------------------------------------------
// Host-side helpers
// ---------------------------------------------------------------------------
static void cvt(const float* x, f16* y, long long n, hipStream_t s) {
  int thr = 256;
  unsigned blocks = (unsigned)((n + thr - 1) / thr);
  cvt_f16_kernel<<<blocks, thr, 0, s>>>(x, y, n);
}

static void launch_gemm(bool transB, const f16* A, const f16* Bm,
                        const float* bias, float* Cf, f16* Ch,
                        int M, int N, int K, int lda, int ldb, int ldc,
                        long long sAo, long long sAi, long long sBo,
                        long long sBi, long long sCo, long long sCi,
                        int inner, int batches, float alpha, int act,
                        hipStream_t s) {
  dim3 grid((M + 63) / 64, (N + 63) / 64, batches);
  dim3 block(128);
  if (transB)
    gemm_wmma<true><<<grid, block, 0, s>>>(A, Bm, bias, Cf, Ch, M, N, K, lda,
                                           ldb, ldc, sAo, sAi, sBo, sBi, sCo,
                                           sCi, inner, alpha, act);
  else
    gemm_wmma<false><<<grid, block, 0, s>>>(A, Bm, bias, Cf, Ch, M, N, K, lda,
                                            ldb, ldc, sAo, sAi, sBo, sBi, sCo,
                                            sCi, inner, alpha, act);
}

// ---------------------------------------------------------------------------
extern "C" void kernel_launch(void* const* d_in, const int* in_sizes, int n_in,
                              void* d_out, int out_size, void* d_ws,
                              size_t ws_size, hipStream_t stream) {
  (void)in_sizes; (void)n_in; (void)out_size; (void)ws_size;
  const float* emb    = (const float*)d_in[0];
  const float* amask  = (const float*)d_in[1];
  const float* eln_s  = (const float*)d_in[2];
  const float* eln_b  = (const float*)d_in[3];
  const float* relemb = (const float*)d_in[4];
  const float* rln_s  = (const float*)d_in[5];
  const float* rln_b  = (const float*)d_in[6];
  const float* Wq = (const float*)d_in[7];  const float* bq = (const float*)d_in[8];
  const float* Wk = (const float*)d_in[9];  const float* bk = (const float*)d_in[10];
  const float* Wv = (const float*)d_in[11]; const float* bv = (const float*)d_in[12];
  const float* Wo = (const float*)d_in[13]; const float* bo = (const float*)d_in[14];
  const float* aln_s = (const float*)d_in[15]; const float* aln_b = (const float*)d_in[16];
  const float* Wi = (const float*)d_in[17]; const float* bi = (const float*)d_in[18];
  const float* Wf = (const float*)d_in[19]; const float* bf = (const float*)d_in[20];
  const float* fln_s = (const float*)d_in[21]; const float* fln_b = (const float*)d_in[22];
  const float* Wc = (const float*)d_in[23]; const float* bc = (const float*)d_in[24];
  float* out = (float*)d_out;

  // ---- bump allocator over d_ws
  size_t off = 0;
  auto alloc = [&](size_t bytes) -> void* {
    off = (off + 255) & ~(size_t)255;
    void* p = (char*)d_ws + off;
    off += bytes;
    return p;
  };

  const long long nWqkv = (long long)KNL * KHID * KHID;
  const long long nWff  = (long long)KNL * KHID * KFF;
  f16* wq16 = (f16*)alloc(sizeof(f16) * nWqkv);
  f16* wk16 = (f16*)alloc(sizeof(f16) * nWqkv);
  f16* wv16 = (f16*)alloc(sizeof(f16) * nWqkv);
  f16* wo16 = (f16*)alloc(sizeof(f16) * nWqkv);
  f16* wi16 = (f16*)alloc(sizeof(f16) * nWff);
  f16* wf16 = (f16*)alloc(sizeof(f16) * nWff);
  f16* wc16 = (f16*)alloc(sizeof(f16) * (size_t)KHID * KVOCAB);

  const long long nH = (long long)KB * KS * KHID;       // 2,097,152
  float* hF   = (float*)alloc(sizeof(float) * nH);
  f16*   h16  = (f16*)alloc(sizeof(f16) * nH);
  f16*   rel16= (f16*)alloc(sizeof(f16) * (size_t)2 * KSPAN * KHID);
  f16*   q16  = (f16*)alloc(sizeof(f16) * nH);
  f16*   k16  = (f16*)alloc(sizeof(f16) * nH);
  f16*   v16  = (f16*)alloc(sizeof(f16) * nH);
  f16*   pk16 = (f16*)alloc(sizeof(f16) * (size_t)2 * KSPAN * KHID);
  f16*   pq16 = (f16*)alloc(sizeof(f16) * (size_t)2 * KSPAN * KHID);
  const long long nAtt = (long long)KB * KHEADS * KS * KS;  // 16,777,216
  float* scores = (float*)alloc(sizeof(float) * nAtt);
  float* c2p    = (float*)alloc(sizeof(float) * nAtt);
  float* p2c    = (float*)alloc(sizeof(float) * nAtt);
  f16*   probs16= (f16*)alloc(sizeof(f16) * nAtt);
  f16*   ctx16  = (f16*)alloc(sizeof(f16) * nH);
  float* tmpF   = (float*)alloc(sizeof(float) * nH);
  f16*   mid16  = (f16*)alloc(sizeof(f16) * (size_t)KB * KS * KFF);
  int*   c2pI   = (int*)alloc(sizeof(int) * (size_t)KS * KS);
  int*   p2cI   = (int*)alloc(sizeof(int) * (size_t)KS * KS);

  const float inv_scale = 1.0f / sqrtf((float)KDH * 3.0f);  // 1/sqrt(192)

  // ---- weight conversion f32 -> f16
  cvt(Wq, wq16, nWqkv, stream);
  cvt(Wk, wk16, nWqkv, stream);
  cvt(Wv, wv16, nWqkv, stream);
  cvt(Wo, wo16, nWqkv, stream);
  cvt(Wi, wi16, nWff, stream);
  cvt(Wf, wf16, nWff, stream);
  cvt(Wc, wc16, (long long)KHID * KVOCAB, stream);

  // ---- relative-position bucket index tables
  relidx_kernel<<<(KS * KS + 255) / 256, 256, 0, stream>>>(c2pI, p2cI);

  // ---- h = LN(emb) * mask ; rel_e = LN(rel_emb)
  ln_kernel<<<KB * KS, 256, 0, stream>>>(emb, nullptr, eln_s, eln_b, amask,
                                         KHID, hF, h16);
  ln_kernel<<<2 * KSPAN, 256, 0, stream>>>(relemb, nullptr, rln_s, rln_b,
                                           nullptr, KHID, nullptr, rel16);

  const int M = KB * KS;  // 4096
  for (int l = 0; l < KNL; ++l) {
    const f16* wq_l = wq16 + (long long)l * KHID * KHID;
    const f16* wk_l = wk16 + (long long)l * KHID * KHID;
    const f16* wv_l = wv16 + (long long)l * KHID * KHID;
    const f16* wo_l = wo16 + (long long)l * KHID * KHID;
    const f16* wi_l = wi16 + (long long)l * KHID * KFF;
    const f16* wf_l = wf16 + (long long)l * KFF * KHID;

    // Q, K, V projections (f16 out)
    launch_gemm(false, h16, wq_l, bq + l * KHID, nullptr, q16, M, KHID, KHID,
                KHID, KHID, KHID, 0, 0, 0, 0, 0, 0, 1, 1, 1.f, 0, stream);
    launch_gemm(false, h16, wk_l, bk + l * KHID, nullptr, k16, M, KHID, KHID,
                KHID, KHID, KHID, 0, 0, 0, 0, 0, 0, 1, 1, 1.f, 0, stream);
    launch_gemm(false, h16, wv_l, bv + l * KHID, nullptr, v16, M, KHID, KHID,
                KHID, KHID, KHID, 0, 0, 0, 0, 0, 0, 1, 1, 1.f, 0, stream);
    // shared pos projections: pk = rel_e@Wk + bk, pq = rel_e@Wq + bq
    launch_gemm(false, rel16, wk_l, bk + l * KHID, nullptr, pk16, 2 * KSPAN,
                KHID, KHID, KHID, KHID, KHID, 0, 0, 0, 0, 0, 0, 1, 1, 1.f, 0,
                stream);
    launch_gemm(false, rel16, wq_l, bq + l * KHID, nullptr, pq16, 2 * KSPAN,
                KHID, KHID, KHID, KHID, KHID, 0, 0, 0, 0, 0, 0, 1, 1, 1.f, 0,
                stream);

    // scores = (Q @ K^T) / SCALE     batched over (b,h), K=DH=64
    launch_gemm(true, q16, k16, nullptr, scores, nullptr, KS, KS, KDH, KHID,
                KHID, KS, (long long)KS * KHID, KDH, (long long)KS * KHID, KDH,
                (long long)KHEADS * KS * KS, (long long)KS * KS, KHEADS,
                KB * KHEADS, inv_scale, 0, stream);
    // c2p_full = (Q @ pk^T) / SCALE   [bh, S, 2*SPAN]
    launch_gemm(true, q16, pk16, nullptr, c2p, nullptr, KS, 2 * KSPAN, KDH,
                KHID, KHID, 2 * KSPAN, (long long)KS * KHID, KDH, 0, KDH,
                (long long)KHEADS * KS * 2 * KSPAN, (long long)KS * 2 * KSPAN,
                KHEADS, KB * KHEADS, inv_scale, 0, stream);
    // p2c_full = (K @ pq^T) / SCALE   [bh, S, 2*SPAN]
    launch_gemm(true, k16, pq16, nullptr, p2c, nullptr, KS, 2 * KSPAN, KDH,
                KHID, KHID, 2 * KSPAN, (long long)KS * KHID, KDH, 0, KDH,
                (long long)KHEADS * KS * 2 * KSPAN, (long long)KS * 2 * KSPAN,
                KHEADS, KB * KHEADS, inv_scale, 0, stream);

    // gather + masked softmax -> probs (f16)
    softmax_gather_kernel<<<KB * KHEADS * KS, 256, 0, stream>>>(
        scores, c2p, p2c, c2pI, p2cI, amask, probs16);

    // ctx = probs @ V   (batched, N=DH=64, f16 out interleaved [B,S,H,DH])
    launch_gemm(false, probs16, v16, nullptr, nullptr, ctx16, KS, KDH, KS, KS,
                KHID, KHID, (long long)KHEADS * KS * KS, (long long)KS * KS,
                (long long)KS * KHID, KDH, (long long)KS * KHID, KDH, KHEADS,
                KB * KHEADS, 1.f, 0, stream);

    // o = ctx @ Wo + bo ; h = LN(o + h)
    launch_gemm(false, ctx16, wo_l, bo + l * KHID, tmpF, nullptr, M, KHID,
                KHID, KHID, KHID, KHID, 0, 0, 0, 0, 0, 0, 1, 1, 1.f, 0, stream);
    ln_kernel<<<M, 256, 0, stream>>>(tmpF, hF, aln_s + l * KHID,
                                     aln_b + l * KHID, nullptr, KHID, hF, h16);

    // f = gelu(h @ Wi + bi)  (exact erf GELU fused in epilogue)
    launch_gemm(false, h16, wi_l, bi + l * KFF, nullptr, mid16, M, KFF, KHID,
                KHID, KFF, KFF, 0, 0, 0, 0, 0, 0, 1, 1, 1.f, 1, stream);
    // h = LN(f @ Wf + bf + h)
    launch_gemm(false, mid16, wf_l, bf + l * KHID, tmpF, nullptr, M, KHID, KFF,
                KFF, KHID, KHID, 0, 0, 0, 0, 0, 0, 1, 1, 1.f, 0, stream);
    ln_kernel<<<M, 256, 0, stream>>>(tmpF, hF, fln_s + l * KHID,
                                     fln_b + l * KHID, nullptr, KHID, hF, h16);
  }

  // logits = h @ Wc + bc   (N=100, edge-guarded in kernel)
  launch_gemm(false, h16, wc16, bc, out, nullptr, M, KVOCAB, KHID, KHID,
              KVOCAB, KVOCAB, 0, 0, 0, 0, 0, 0, 1, 1, 1.f, 0, stream);
}
